// linearGPT2cell_19602230739132
// MI455X (gfx1250) — matrix operand order repaired and checked by reading, not compiled, not run on hardware
//
#include <hip/hip_runtime.h>
#include <math.h>

// ---------------------------------------------------------------------------
// Types for CDNA5 WMMA (wave32): v16bf A/B fragments, v8f accumulator.
// ---------------------------------------------------------------------------
typedef __attribute__((ext_vector_type(16))) __bf16          v16bf;
typedef __attribute__((ext_vector_type(8)))  float           v8f;
typedef __attribute__((ext_vector_type(8)))  unsigned short  v8us;
typedef __attribute__((ext_vector_type(4)))  int             v4i;

struct US16 { v8us lo; v8us hi; };   // 32 bytes, bit-castable to v16bf

// AS-qualified int4 for the async-to-LDS builtin:
//   (v4i addrspace(1)* src_global, v4i addrspace(3)* dst_lds, imm off, imm cpol)
typedef __attribute__((address_space(1))) v4i g_v4i;   // global
typedef __attribute__((address_space(3))) v4i l_v4i;   // LDS

// Async global->LDS copy (CDNA5 GLOBAL_LOAD_ASYNC_TO_LDS_B128, ASYNCcnt).
#if defined(__has_builtin)
#  if __has_builtin(__builtin_amdgcn_global_load_async_to_lds_b128) && \
      __has_builtin(__builtin_amdgcn_s_wait_asynccnt)
#    define USE_ASYNC_LDS 1
#  endif
#endif
#ifndef USE_ASYNC_LDS
#  define USE_ASYNC_LDS 0
#endif

static __device__ __forceinline__ void async_cp32B(const unsigned short* g, unsigned short* l) {
#if USE_ASYNC_LDS
    g_v4i* gp = (g_v4i*)(size_t)g;
    l_v4i* lp = (l_v4i*)(unsigned)(size_t)l;   // generic LDS addr: low 32 bits = LDS offset
    __builtin_amdgcn_global_load_async_to_lds_b128(gp, lp, 0, 0);
    __builtin_amdgcn_global_load_async_to_lds_b128(gp, lp, 16, 0);
#else
    (void)g; (void)l;
#endif
}

static __device__ __forceinline__ unsigned short f2bf(float f) {
    unsigned u = __builtin_bit_cast(unsigned, f);
    unsigned r = u + 0x7FFFu + ((u >> 16) & 1u);   // round-to-nearest-even
    return (unsigned short)(r >> 16);
}

// ---------------------------------------------------------------------------
// Problem constants
// ---------------------------------------------------------------------------
#define BSZ 2048
#define DIM 1024
#define NH  16
#define DH  64
#define RF  32

// ---------------------------------------------------------------------------
// Weight prep: f32 [K,N] -> bf16 bits [N,K]  (transpose via LDS tile)
// grid = (N/32, K/32), block = 256
// ---------------------------------------------------------------------------
__global__ __launch_bounds__(256)
void wtrans_bf16_kernel(const float* __restrict__ src, unsigned short* __restrict__ dst,
                        int K, int N) {
    __shared__ float tile[32][33];
    const int tx = threadIdx.x & 31;
    const int ty = threadIdx.x >> 5;           // 0..7
    const int k0 = blockIdx.y * 32;
    const int n0 = blockIdx.x * 32;
#pragma unroll
    for (int i = 0; i < 4; ++i)
        tile[ty + i * 8][tx] = src[(size_t)(k0 + ty + i * 8) * N + n0 + tx];
    __syncthreads();
#pragma unroll
    for (int i = 0; i < 4; ++i)
        dst[(size_t)(n0 + ty + i * 8) * K + k0 + tx] = f2bf(tile[tx][ty + i * 8]);
}

// ---------------------------------------------------------------------------
// LayerNorm over rows of [*,1024]; optional residual input; writes f32 and/or
// bf16 copies of the result.
// ---------------------------------------------------------------------------
__global__ __launch_bounds__(256)
void ln_kernel(const float* __restrict__ in, const float* __restrict__ res,
               const float* __restrict__ g, const float* __restrict__ bb,
               float* __restrict__ outf, unsigned short* __restrict__ outbf) {
    const int row = blockIdx.x;
    const int t   = threadIdx.x;
    __shared__ float s1[256];
    __shared__ float s2[256];
    float v[4];
    float ls = 0.f, lq = 0.f;
#pragma unroll
    for (int i = 0; i < 4; ++i) {
        int col = t + i * 256;
        float x = in[row * DIM + col];
        if (res) x += res[row * DIM + col];
        v[i] = x; ls += x; lq += x * x;
    }
    s1[t] = ls; s2[t] = lq;
    __syncthreads();
    for (int off = 128; off > 0; off >>= 1) {
        if (t < off) { s1[t] += s1[t + off]; s2[t] += s2[t + off]; }
        __syncthreads();
    }
    const float mean = s1[0] * (1.f / (float)DIM);
    const float var  = s2[0] * (1.f / (float)DIM) - mean * mean;
    const float rstd = rsqrtf(var + 1e-3f);
#pragma unroll
    for (int i = 0; i < 4; ++i) {
        int col = t + i * 256;
        float o = g[col] * (v[i] - mean) * rstd + bb[col];
        if (outf)  outf[row * DIM + col]  = o;
        if (outbf) outbf[row * DIM + col] = f2bf(o);
    }
}

// ---------------------------------------------------------------------------
// bf16 WMMA GEMM: C[M,N] = A[M,K] @ B[K,N], B supplied TRANSPOSED ([N,K]).
// Block: 256 threads = 8 waves (4x2), block tile 128x128,
// wave tile 32x64 (2x4 WMMA accumulators), K staged 32 at a time.
// Double-buffered LDS fed by async global->LDS loads when available.
//   EPI: 0 = +bias -> f32 ; 1 = gelu(+bias) -> bf16 ; 2 = +bias +add -> f32
// N, K are compile-time to keep epilogue address math in shifts/adds.
// ---------------------------------------------------------------------------
#define TILE_MN 128
#define TILE_K  32
#define LDS_STR 40    // 32 + 8 pad (ushorts); row pitch 80B (16B aligned)

static __device__ __forceinline__ void compute_step(
    const unsigned short* __restrict__ As, const unsigned short* __restrict__ Bts,
    int wm, int wn, int l16, int hi, v8f (&acc)[2][4]) {
    v16bf afr[2];
    const int koff = hi ? 8 : 0;   // A: lanes 0-15 K{0..7,16..23}; 16-31 K{8..15,24..31}
#pragma unroll
    for (int t = 0; t < 2; ++t) {
        const unsigned short* p = &As[(wm * 32 + t * 16 + l16) * LDS_STR];
        US16 u;
        u.lo = *reinterpret_cast<const v8us*>(p + koff);
        u.hi = *reinterpret_cast<const v8us*>(p + koff + 16);
        afr[t] = __builtin_bit_cast(v16bf, u);
    }
    v16bf bfr[4];
    const int kb = hi ? 8 : 0;     // B: lanes 0-15 K{0..7,16..23}; 16-31 K{8..15,24..31}
#pragma unroll
    for (int u = 0; u < 4; ++u) {
        const unsigned short* p = &Bts[(wn * 64 + u * 16 + l16) * LDS_STR];
        US16 w;
        w.lo = *reinterpret_cast<const v8us*>(p + kb);
        w.hi = *reinterpret_cast<const v8us*>(p + kb + 16);
        bfr[u] = __builtin_bit_cast(v16bf, w);
    }
#pragma unroll
    for (int t = 0; t < 2; ++t)
#pragma unroll
        for (int u = 0; u < 4; ++u)
            acc[t][u] = __builtin_amdgcn_wmma_f32_16x16x32_bf16(
                false, afr[t], false, bfr[u], (short)0, acc[t][u], false, false);
}

template <int EPI, int N, int K>
__global__ __launch_bounds__(256)
void gemm_bf16_wmma(const unsigned short* __restrict__ A,
                    const unsigned short* __restrict__ Bt,
                    const float* __restrict__ bias,
                    const float* __restrict__ add,
                    float* __restrict__ Cf,
                    unsigned short* __restrict__ Cb) {
#if USE_ASYNC_LDS
    __shared__ __align__(16) unsigned short As [2][TILE_MN * LDS_STR];
    __shared__ __align__(16) unsigned short Bts[2][TILE_MN * LDS_STR];
#else
    __shared__ __align__(16) unsigned short As [1][TILE_MN * LDS_STR];
    __shared__ __align__(16) unsigned short Bts[1][TILE_MN * LDS_STR];
#endif

    const int tid  = threadIdx.x;
    const int wid  = tid >> 5;
    const int lane = tid & 31;
    const int wm   = wid >> 1;         // 0..3  (rows, 32 each)
    const int wn   = wid & 1;          // 0..1  (cols, 64 each)
    const int l16  = lane & 15;
    const int hi   = (lane >> 4) & 1;

    const int tile_m = blockIdx.y * TILE_MN;
    const int tile_n = blockIdx.x * TILE_MN;

    // staging coords: each thread moves 16 ushorts (32B) of each tile
    const int srow = tid >> 1;
    const int scol = (tid & 1) << 4;
    const int soff = srow * LDS_STR + scol;
    const unsigned short* gA = A  + (size_t)(tile_m + srow) * K + scol;
    const unsigned short* gB = Bt + (size_t)(tile_n + srow) * K + scol;

    v8f acc[2][4] = {};

#if USE_ASYNC_LDS
    // prologue: async-stage first K-tile into buffer 0
    async_cp32B(gA, &As[0][soff]);
    async_cp32B(gB, &Bts[0][soff]);
#pragma unroll 2
    for (int kt = 0; kt < K / TILE_K; ++kt) {
        const int cur = kt & 1;
        __builtin_amdgcn_s_wait_asynccnt(0);
        __syncthreads();
        if (kt + 1 < K / TILE_K) {     // overlap next tile's DMA with compute
            async_cp32B(gA + (kt + 1) * TILE_K, &As[cur ^ 1][soff]);
            async_cp32B(gB + (kt + 1) * TILE_K, &Bts[cur ^ 1][soff]);
        }
        compute_step(&As[cur][0], &Bts[cur][0], wm, wn, l16, hi, acc);
    }
#else
    for (int k0 = 0; k0 < K; k0 += TILE_K) {
        // distinct temps -> 4 loads in flight, single wait, then 4 LDS stores
        const v8us* ga = reinterpret_cast<const v8us*>(gA + k0);
        const v8us* gb = reinterpret_cast<const v8us*>(gB + k0);
        v8us a0 = ga[0], a1 = ga[1], b0 = gb[0], b1 = gb[1];
        if (k0 + TILE_K < K) {
            __builtin_prefetch(gA + k0 + TILE_K, 0, 1);
            __builtin_prefetch(gB + k0 + TILE_K, 0, 1);
        }
        v8us* la = reinterpret_cast<v8us*>(&As[0][soff]);
        v8us* lb = reinterpret_cast<v8us*>(&Bts[0][soff]);
        la[0] = a0; la[1] = a1; lb[0] = b0; lb[1] = b1;
        __syncthreads();
        compute_step(&As[0][0], &Bts[0][0], wm, wn, l16, hi, acc);
        __syncthreads();
    }
#endif

    // ---- epilogue: lane -> N column, accumulator index -> M row (+8 upper lanes)
#pragma unroll
    for (int u = 0; u < 4; ++u) {
        const int col = tile_n + wn * 64 + u * 16 + l16;
        const float bv = bias[col];
#pragma unroll
        for (int t = 0; t < 2; ++t) {
            const int row0 = tile_m + wm * 32 + t * 16 + hi * 8;
#pragma unroll
            for (int v = 0; v < 8; ++v) {
                const int row = row0 + v;
                float val = acc[t][u][v] + bv;
                if (EPI == 0) {
                    Cf[(size_t)row * N + col] = val;
                } else if (EPI == 1) {
                    float gl = 0.5f * val * (1.f + erff(val * 0.70710678118654752f));
                    Cb[(size_t)row * N + col] = f2bf(gl);
                } else {
                    Cf[(size_t)row * N + col] = val + add[(size_t)row * N + col];
                }
            }
        }
    }
}

// ---------------------------------------------------------------------------
// Fused linear-attention state update. One 128-thread block per (b,h):
//   phi_{k1,k2,q3,q4} = sigmoid(k/q . w_i[h])         (4 x 32 dots of len 64)
//   new_s = s + phi_k1 (x) v ; num = phi_q3 . new_s   (streamed once)
//   new_z = z + phi_k2       ; den = phi_q4 . new_z
//   head  = num / den  (stored bf16 for the projection GEMM)
// ---------------------------------------------------------------------------
__global__ __launch_bounds__(128)
void state_kernel(const float* __restrict__ qkv,
                  const float* __restrict__ s,
                  const float* __restrict__ z,
                  const float* __restrict__ w1, const float* __restrict__ w2,
                  const float* __restrict__ w3, const float* __restrict__ w4,
                  unsigned short* __restrict__ head_bf,
                  float* __restrict__ new_s, float* __restrict__ new_z) {
    const int b = blockIdx.x >> 4;
    const int h = blockIdx.x & 15;
    const int t = threadIdx.x;

    __shared__ float qv[DH], kv[DH], vv[DH];
    __shared__ float phis[128];
    __shared__ float red[128];
    __shared__ float den_red[RF];
    __shared__ float den_sh;

    const size_t qbase = (size_t)b * (3 * DIM) + h * DH;
    if (t < DH) {
        qv[t] = qkv[qbase + t];
        kv[t] = qkv[qbase + DIM + t];
        vv[t] = qkv[qbase + 2 * DIM + t];
    }
    __syncthreads();

    // 4 random-feature projections + sigmoid
    {
        const int mat = t >> 5, r = t & 31;
        const float* w = (mat == 0 ? w1 : mat == 1 ? w2 : mat == 2 ? w3 : w4)
                         + (size_t)h * (DH * RF) + r;
        const float* src = (mat < 2) ? kv : qv;
        float a = 0.f;
#pragma unroll 8
        for (int d = 0; d < DH; ++d) a += src[d] * w[d * RF];
        phis[t] = 1.f / (1.f + __expf(-a));
    }
    __syncthreads();

    // new_s streaming + partial num  (thread: d = t&63, r-half = t>>6)
    {
        const int d = t & 63, half = t >> 6;
        const size_t base = (size_t)(b * NH + h) * (RF * DH);
        float p3 = 0.f;
        const int r0 = half * 16;
#pragma unroll 4
        for (int r = r0; r < r0 + 16; ++r) {
            const size_t idx = base + (size_t)r * DH + d;
            float ns = s[idx] + phis[r] * vv[d];
            new_s[idx] = ns;
            p3 += phis[64 + r] * ns;
        }
        red[t] = p3;
    }
    __syncthreads();

    float num_d = 0.f;
    if (t < 64) num_d = red[t] + red[t + 64];
    if (t < RF) {
        const size_t zi = (size_t)b * (RF * NH) + h * RF + t;
        float nz = z[zi] + phis[32 + t];
        new_z[zi] = nz;
        den_red[t] = phis[96 + t] * nz;
    }
    __syncthreads();
    if (t == 0) {
        float ds = 0.f;
        for (int r = 0; r < RF; ++r) ds += den_red[r];
        den_sh = ds;
    }
    __syncthreads();
    if (t < DH) head_bf[(size_t)b * DIM + h * DH + t] = f2bf(num_d / den_sh);
}

// ---------------------------------------------------------------------------
// Host launcher
// ---------------------------------------------------------------------------
extern "C" void kernel_launch(void* const* d_in, const int* in_sizes, int n_in,
                              void* d_out, int out_size, void* d_ws, size_t ws_size,
                              hipStream_t stream) {
    (void)in_sizes; (void)n_in; (void)out_size; (void)ws_size;

    const float* x      = (const float*)d_in[0];
    const float* s      = (const float*)d_in[1];
    const float* z      = (const float*)d_in[2];
    const float* ln1_g  = (const float*)d_in[3];
    const float* ln1_b  = (const float*)d_in[4];
    const float* ln2_g  = (const float*)d_in[5];
    const float* ln2_b  = (const float*)d_in[6];
    const float* W_attn = (const float*)d_in[7];
    const float* b_attn = (const float*)d_in[8];
    const float* W_proj = (const float*)d_in[9];
    const float* b_proj = (const float*)d_in[10];
    const float* W_fc   = (const float*)d_in[11];
    const float* b_fc   = (const float*)d_in[12];
    const float* W_mp   = (const float*)d_in[13];
    const float* b_mp   = (const float*)d_in[14];
    const float* w1     = (const float*)d_in[15];
    const float* w2     = (const float*)d_in[16];
    const float* w3     = (const float*)d_in[17];
    const float* w4     = (const float*)d_in[18];

    // d_out = [ out (B*D) | new_s (B*D*R) | new_z (B*R*H) ]  (f32)
    float* out_f = (float*)d_out;
    float* new_s = out_f + (size_t)BSZ * DIM;
    float* new_z = new_s + (size_t)BSZ * DIM * RF;

    // workspace carve-up (256B aligned)
    char* ws = (char*)d_ws;
    size_t off = 0;
    auto carve = [&](size_t bytes) -> char* {
        char* p = ws + off;
        off += (bytes + 255) & ~(size_t)255;
        return p;
    };
    unsigned short* h1_bf   = (unsigned short*)carve((size_t)BSZ * DIM * 2);
    unsigned short* WattnT  = (unsigned short*)carve((size_t)DIM * 3 * DIM * 2);
    float*          qkv     = (float*)carve((size_t)BSZ * 3 * DIM * 4);
    unsigned short* head_bf = (unsigned short*)carve((size_t)BSZ * DIM * 2);
    unsigned short* WprojT  = (unsigned short*)carve((size_t)DIM * DIM * 2);
    float*          attn    = (float*)carve((size_t)BSZ * DIM * 4);
    float*          h2_f    = (float*)carve((size_t)BSZ * DIM * 4);
    unsigned short* h2_bf   = (unsigned short*)carve((size_t)BSZ * DIM * 2);
    unsigned short* WfcT    = (unsigned short*)carve((size_t)DIM * 4 * DIM * 2);
    unsigned short* act_bf  = (unsigned short*)carve((size_t)BSZ * 4 * DIM * 2);
    unsigned short* WmpT    = (unsigned short*)carve((size_t)4 * DIM * DIM * 2);

    // 1) weights f32 [K,N] -> bf16 transposed [N,K]
    wtrans_bf16_kernel<<<dim3(3 * DIM / 32, DIM / 32),     256, 0, stream>>>(W_attn, WattnT, DIM,     3 * DIM);
    wtrans_bf16_kernel<<<dim3(DIM / 32,     DIM / 32),     256, 0, stream>>>(W_proj, WprojT, DIM,     DIM);
    wtrans_bf16_kernel<<<dim3(4 * DIM / 32, DIM / 32),     256, 0, stream>>>(W_fc,   WfcT,   DIM,     4 * DIM);
    wtrans_bf16_kernel<<<dim3(DIM / 32,     4 * DIM / 32), 256, 0, stream>>>(W_mp,   WmpT,   4 * DIM, DIM);

    // 2) h1 = LN1(x) -> bf16
    ln_kernel<<<BSZ, 256, 0, stream>>>(x, nullptr, ln1_g, ln1_b, nullptr, h1_bf);

    // 3) qkv = h1 @ W_attn + b_attn          [2048 x 3072], K=1024
    gemm_bf16_wmma<0, 3 * DIM, DIM><<<dim3(3 * DIM / TILE_MN, BSZ / TILE_MN), 256, 0, stream>>>(
        h1_bf, WattnT, b_attn, nullptr, qkv, nullptr);

    // 4) fused state update: new_s, new_z, head
    state_kernel<<<BSZ * NH, 128, 0, stream>>>(qkv, s, z, w1, w2, w3, w4,
                                               head_bf, new_s, new_z);

    // 5) attention = head @ W_proj + b_proj  [2048 x 1024], K=1024
    gemm_bf16_wmma<0, DIM, DIM><<<dim3(DIM / TILE_MN, BSZ / TILE_MN), 256, 0, stream>>>(
        head_bf, WprojT, b_proj, nullptr, attn, nullptr);

    // 6) h2 = LN2(attention + x) -> f32 + bf16
    ln_kernel<<<BSZ, 256, 0, stream>>>(attn, x, ln2_g, ln2_b, h2_f, h2_bf);

    // 7) act = gelu(h2 @ W_fc + b_fc) -> bf16  [2048 x 4096], K=1024
    gemm_bf16_wmma<1, 4 * DIM, DIM><<<dim3(4 * DIM / TILE_MN, BSZ / TILE_MN), 256, 0, stream>>>(
        h2_bf, WfcT, b_fc, nullptr, nullptr, act_bf);

    // 8) out = h2 + act @ W_mp + b_mp        [2048 x 1024], K=4096
    gemm_bf16_wmma<2, DIM, 4 * DIM><<<dim3(DIM / TILE_MN, BSZ / TILE_MN), 256, 0, stream>>>(
        act_bf, WmpT, b_mp, h2_f, out_f, nullptr);
}
